// CSELoss_24103356465211
// MI455X (gfx1250) — compile-verified, matrix operand order checked by reading, not run on hardware
//
#include <hip/hip_runtime.h>
#include <stdint.h>

// ---------------------------------------------------------------------------
// CDNA5 (gfx1250, wave32) contrastive-similarity loss.
//  Pass 1: row-normalize + bf16 hi/lo precision split  -> workspace
//  Pass 2: WMMA x·x^T, B tiles staged via GLOBAL_LOAD_ASYNC_TO_LDS (double
//          buffered), fragments read back as ds_load_b128, 3 independent
//          WMMA accumulator chains, bounded-max streaming exp-sum per row
//          (cos<=1 -> u<=1/alpha, single pass, no N^2 materialization)
//  Pass 3: deterministic final reduction -> scalar loss
// ---------------------------------------------------------------------------

typedef __attribute__((ext_vector_type(16))) __bf16 v16bf;
typedef __attribute__((ext_vector_type(8)))  float  v8f;

#define DIMS      128
#define BM        64
#define ROWPAD    144      // LDS row stride in bf16 elems (288 B, 16B-aligned)
#define ALPHA_INV 20.0f
// K1 = log2(e)/alpha ; also equals C*log2(e) with C = 1/alpha
#define K1C       28.853900817779268f
#define LN2F      0.6931471805599453f

union Frag {
  uint4 q[2];
  v16bf v;
};

static __device__ __forceinline__ v8f wmma_bf16(v16bf a, v16bf b, v8f c) {
  return __builtin_amdgcn_wmma_f32_16x16x32_bf16(false, a, false, b,
                                                 (short)0, c, false, false);
}

static __device__ __forceinline__ uint16_t f32_to_bf16_rne(float x) {
  uint32_t u = __builtin_bit_cast(uint32_t, x);
  u += 0x7FFFu + ((u >> 16) & 1u);
  return (uint16_t)(u >> 16);
}
static __device__ __forceinline__ float bf16_bits_to_f32(uint16_t h) {
  uint32_t u = ((uint32_t)h) << 16;
  return __builtin_bit_cast(float, u);
}

// ---------------------------------------------------------------------------
// Pass 1: one block per row (128 threads).
// ---------------------------------------------------------------------------
__global__ void __launch_bounds__(DIMS)
norm_split_kernel(const float* __restrict__ rep,
                  uint16_t* __restrict__ xhi,
                  uint16_t* __restrict__ xlo) {
  const int row = blockIdx.x;
  const int t   = threadIdx.x;
  const float v = rep[(size_t)row * DIMS + t];
  float ss = v * v;
#pragma unroll
  for (int msk = 16; msk >= 1; msk >>= 1) ss += __shfl_xor(ss, msk, 32);
  __shared__ float ws[4];
  if ((t & 31) == 0) ws[t >> 5] = ss;
  __syncthreads();
  const float tot = ws[0] + ws[1] + ws[2] + ws[3];
  const float inv = 1.0f / fmaxf(sqrtf(tot), 1e-8f);
  const float x   = v * inv;
  const uint16_t hb = f32_to_bf16_rne(x);
  const float    lo = x - bf16_bits_to_f32(hb);
  xhi[(size_t)row * DIMS + t] = hb;
  xlo[(size_t)row * DIMS + t] = f32_to_bf16_rne(lo);
}

// ---------------------------------------------------------------------------
// Async-stage one 32-column B tile (hi rows 0..31, lo rows 32..63) into LDS.
// 16 KB per tile = 1024 x 16B chunks; 256 threads x 4 async B128 issues.
// ldsBase is the group-relative LDS byte offset of the destination buffer
// (exactly what the VDST operand of global_load_async_to_lds expects).
// ---------------------------------------------------------------------------
static __device__ __forceinline__ void stage_tile(const uint16_t* __restrict__ xhi,
                                                  const uint16_t* __restrict__ xlo,
                                                  uint32_t ldsBase, int j, int tid) {
#pragma unroll
  for (int i = 0; i < 4; ++i) {
    const int c    = tid + i * 256;                 // 0..1023
    const int isLo = c >> 9;                        // 512 chunks hi, 512 lo
    const int cc   = c & 511;
    const int row  = cc >> 4;                       // 0..31
    const int ch   = cc & 15;                       // 16B chunk within row
    const uint16_t* src = (isLo ? xlo : xhi) + (size_t)(j + row) * DIMS + ch * 8;
    const uint32_t  dst = ldsBase +
        (uint32_t)(((row + (isLo << 5)) * ROWPAD + ch * 8) * 2);
    const uint64_t  ga  = (uint64_t)(uintptr_t)src;
    asm volatile("global_load_async_to_lds_b128 %0, %1, off"
                 :: "v"(dst), "v"(ga) : "memory");
  }
}

static __device__ __forceinline__ void async_wait_barrier() {
  asm volatile("s_wait_asynccnt 0x0" ::: "memory");
  __syncthreads();
}

// ---------------------------------------------------------------------------
// Pass 2: workgroup = 8 waves = 64 rows; wave w owns tile
// (row_sub = w>>1, col_sub = w&1) and sweeps all N columns in 32-col steps.
// ---------------------------------------------------------------------------
__global__ void __launch_bounds__(256)
simloss_kernel(const uint16_t* __restrict__ xhi,
               const uint16_t* __restrict__ xlo,
               float* __restrict__ partials, int N) {
  const int wave    = threadIdx.x >> 5;
  const int lane    = threadIdx.x & 31;
  const int row_sub = wave >> 1;
  const int col_sub = wave & 1;
  const int rb      = blockIdx.x * BM;
  const int m       = lane & 15;   // frag row/col within 16-subtile; C/D col
  const int h       = lane >> 4;   // half-wave selector
  const int kb      = h * 8;       // K sub-offset (ISA 16-bit frag layout)

  __shared__ __align__(16) uint16_t sB[2][BM][ROWPAD];   // double-buffered tile
  __shared__ float lds_S[2][BM];
  __shared__ float lds_T[BM];
  if (threadIdx.x < BM) lds_T[threadIdx.x] = 0.0f;

  // Hoist this wave's A fragments: 16 rows x K=128, hi and lo (global, once).
  v16bf a_hi[4], a_lo[4];
  {
    const size_t arow = (size_t)(rb + row_sub * 16 + m) * DIMS;
#pragma unroll
    for (int s = 0; s < 4; ++s) {
      const int K0 = s * 32 + kb;
      Frag f;
      f.q[0] = *(const uint4*)(xhi + arow + K0);
      f.q[1] = *(const uint4*)(xhi + arow + K0 + 16);
      a_hi[s] = f.v;
      f.q[0] = *(const uint4*)(xlo + arow + K0);
      f.q[1] = *(const uint4*)(xlo + arow + K0 + 16);
      a_lo[s] = f.v;
    }
  }

  // Group-relative LDS offsets for the async-DMA destination operands only.
  const uint32_t ldsB0 = (uint32_t)(uintptr_t)&sB[0][0][0];
  const uint32_t ldsB1 = (uint32_t)(uintptr_t)&sB[1][0][0];

  // Prologue: stage tile j=0 into buffer 0, wait, barrier.
  stage_tile(xhi, xlo, ldsB0, 0, threadIdx.x);
  async_wait_barrier();

  float ssum[8];
#pragma unroll
  for (int r = 0; r < 8; ++r) ssum[r] = 0.0f;
  const int growbase = rb + row_sub * 16 + 8 * h;
  const int brow     = col_sub * 16 + m;            // staged row for B frags

  for (int j = 0; j < N; j += 32) {
    const int buf = (j >> 5) & 1;
    if (j + 32 < N) stage_tile(xhi, xlo, buf ? ldsB0 : ldsB1, j + 32, threadIdx.x);

    // Fragment reads derived straight from the __shared__ object -> ds_load_b128
    // (DScnt only; no FLAT/aperture cost).
    const uint16_t* bh_p = &sB[buf][brow][0];
    const uint16_t* bl_p = &sB[buf][brow + 32][0];

    // Three independent accumulator chains -> no serialized WMMA pipeline.
    v8f acc_hh = {}, acc_hl = {}, acc_lh = {};
#pragma unroll
    for (int s = 0; s < 4; ++s) {
      const int K0 = s * 32 + kb;
      Frag bh, bl;
      bh.q[0] = *(const uint4*)(bh_p + K0);
      bh.q[1] = *(const uint4*)(bh_p + K0 + 16);
      bl.q[0] = *(const uint4*)(bl_p + K0);
      bl.q[1] = *(const uint4*)(bl_p + K0 + 16);
      acc_hh = wmma_bf16(a_hi[s], bh.v, acc_hh);
      acc_hl = wmma_bf16(a_hi[s], bl.v, acc_hl);
      acc_lh = wmma_bf16(a_lo[s], bh.v, acc_lh);
    }
    const v8f vv = (acc_hh + acc_hl) + acc_lh;

    const int tilecol = j + col_sub * 16;
    if ((unsigned)(tilecol - rb) < (unsigned)BM) {
      // Rare path (2 of 256 steps/wave): diagonal + target column live here.
      const int gcol = tilecol + m;
#pragma unroll
      for (int r = 0; r < 8; ++r) {
        const int   grow = growbase + r;
        const float v    = vv[r];
        const float e = __builtin_amdgcn_exp2f(__builtin_fmaf(v, K1C, -K1C));
        if (gcol != grow) ssum[r] += e;                           // mask diagonal
        if (gcol == (grow ^ 1)) lds_T[grow - rb] = v * ALPHA_INV; // target logit
      }
    } else {
      // Hot path: pure fma + exp2 + add, no compares, no EXEC churn.
#pragma unroll
      for (int r = 0; r < 8; ++r)
        ssum[r] += __builtin_amdgcn_exp2f(__builtin_fmaf(vv[r], K1C, -K1C));
    }

    // My async issues done + everyone finished reading sB[buf].
    async_wait_barrier();
  }

  // Sum across the 16 lanes of each half-wave (columns of the tile).
#pragma unroll
  for (int r = 0; r < 8; ++r) {
    float s = ssum[r];
    s += __shfl_xor(s, 1, 32);
    s += __shfl_xor(s, 2, 32);
    s += __shfl_xor(s, 4, 32);
    s += __shfl_xor(s, 8, 32);
    if (m == 0) lds_S[col_sub][row_sub * 16 + 8 * h + r] = s;
  }
  __syncthreads();

  // Per-row loss term: (1/alpha) + ln(S) - u_tgt ; fixed-order block reduce.
  float part = 0.0f;
  if (threadIdx.x < BM) {
    const float S = lds_S[0][threadIdx.x] + lds_S[1][threadIdx.x];
    part = ALPHA_INV + LN2F * __builtin_amdgcn_logf(S) - lds_T[threadIdx.x];
  }
#pragma unroll
  for (int msk = 16; msk >= 1; msk >>= 1) part += __shfl_xor(part, msk, 32);
  __shared__ float wred[8];
  if (lane == 0) wred[wave] = part;
  __syncthreads();
  if (threadIdx.x == 0) {
    float t = 0.0f;
#pragma unroll
    for (int i = 0; i < 8; ++i) t += wred[i];
    partials[blockIdx.x] = t;
  }
}

// ---------------------------------------------------------------------------
// Pass 3: single wave sums per-block partials in fixed order -> mean loss.
// ---------------------------------------------------------------------------
__global__ void __launch_bounds__(32)
finalize_kernel(const float* __restrict__ partials, int nblocks, float invN,
                float* __restrict__ out) {
  float s = 0.0f;
  for (int i = threadIdx.x; i < nblocks; i += 32) s += partials[i];
#pragma unroll
  for (int msk = 16; msk >= 1; msk >>= 1) s += __shfl_xor(s, msk, 32);
  if (threadIdx.x == 0) out[0] = s * invN;
}

extern "C" void kernel_launch(void* const* d_in, const int* in_sizes, int n_in,
                              void* d_out, int out_size, void* d_ws, size_t ws_size,
                              hipStream_t stream) {
  const float* rep = (const float*)d_in[0];
  const int N = in_sizes[0] / DIMS;          // 8192

  uint16_t* xhi      = (uint16_t*)d_ws;
  uint16_t* xlo      = xhi + (size_t)N * DIMS;
  float*    partials = (float*)(xlo + (size_t)N * DIMS);

  norm_split_kernel<<<N, DIMS, 0, stream>>>(rep, xhi, xlo);

  const int nblocks = N / BM;                // 128 workgroups
  simloss_kernel<<<nblocks, 256, 0, stream>>>(xhi, xlo, partials, N);

  finalize_kernel<<<1, 32, 0, stream>>>(partials, nblocks, 1.0f / (float)N,
                                        (float*)d_out);
}
